// TransformerTranslator_52931176956235
// MI455X (gfx1250) — matrix operand order repaired and checked by reading, not compiled, not run on hardware
//
#include <hip/hip_runtime.h>
#include <hip/hip_bf16.h>
#include <stdint.h>
#include <math.h>

// ---------------------------------------------------------------------------
// Transformer translator forward (fp32 master, f16 WMMA matmuls, fp32 accum).
// gfx1250: wave32, v_wmma_f32_16x16x32_f16 for every GEMM, float4 (b128)
// global loads, global_prefetch for K-pipelining.  NT/NN variants are
// compile-time specializations so the hot NT loop carries no dead pointers.
// ---------------------------------------------------------------------------

typedef __attribute__((ext_vector_type(16))) _Float16 v16h;
typedef __attribute__((ext_vector_type(8)))  float    v8f;

#define D_       512
#define H_       8
#define DK_      64
#define FF_      2048
#define S_       512
#define B_       4
#define BH_      (B_ * H_)
#define NTOK_    (B_ * S_)          /* 2048 rows */
#define TGTV_    32000
#define INV_SCALE_ 0.125f           /* 1/sqrt(64) */
#define EMB_SCALE_ 22.62741699796952f /* sqrt(512) */

// Per-(b,h) sub-matrix offset inside a [B*S, D] activation buffer.
__device__ __forceinline__ int64_t head_off(int z, int ld) {
  int b = z / H_;
  int h = z % H_;
  return (int64_t)b * S_ * ld + (int64_t)h * DK_;
}

__device__ __forceinline__ void cvt8(v16h& f, int base, float4 a, float4 b) {
  f[base + 0] = (_Float16)a.x; f[base + 1] = (_Float16)a.y;
  f[base + 2] = (_Float16)a.z; f[base + 3] = (_Float16)a.w;
  f[base + 4] = (_Float16)b.x; f[base + 5] = (_Float16)b.y;
  f[base + 6] = (_Float16)b.z; f[base + 7] = (_Float16)b.w;
}

// A fragment 16x32 f16: lanes 0-15 hold rows, K 0..15;
// lanes 16-31 same rows, K 16..31.  All runs 16B-aligned -> b128.
__device__ __forceinline__ v16h load_frag_a(const float* __restrict__ p0,
                                            int ld, int row0, int k0) {
  int lane = threadIdx.x & 31;
  const float* p = p0 + (int64_t)(row0 + (lane & 15)) * ld + k0 + ((lane >> 4) ? 16 : 0);
  const float4* q = (const float4*)p;
  float4 f0 = q[0], f1 = q[1];      // K +0..7
  float4 f2 = q[2], f3 = q[3];      // K +8..15
  v16h f;
  cvt8(f, 0, f0, f1);
  cvt8(f, 8, f2, f3);
  return f;
}

// B fragment 32x16 f16, from row-major W[N,K] (computing X @ W^T):
// lane n (0-15) = column N=n, K k0..k0+15; lanes 16-31 same column, K +16.
__device__ __forceinline__ v16h load_frag_b_nt(const float* __restrict__ p0,
                                               int ld, int col0, int k0) {
  int lane = threadIdx.x & 31;
  const float* p = p0 + (int64_t)(col0 + (lane & 15)) * ld + k0 + ((lane >> 4) ? 16 : 0);
  const float4* q = (const float4*)p;
  float4 f0 = q[0], f1 = q[1], f2 = q[2], f3 = q[3];
  v16h f;
  cvt8(f, 0, f0, f1);
  cvt8(f, 8, f2, f3);
  return f;
}

// B fragment from row-major B[K,N] (normal orientation, e.g. P @ V).
__device__ __forceinline__ v16h load_frag_b_nn(const float* __restrict__ p0,
                                               int ld, int col0, int k0) {
  int lane = threadIdx.x & 31;
  int c  = col0 + (lane & 15);
  int kb = k0 + ((lane >> 4) ? 16 : 0);
  v16h f;
#pragma unroll
  for (int i = 0; i < 16; ++i) f[i] = (_Float16)p0[(int64_t)(kb + i) * ld + c];
  return f;
}

// Generic batched WMMA GEMM: C = act(alpha * A@B(^T) + bias).
// 128 threads = 4 waves; block tile 128(M)x64(N); wave tile 32x64
// (2x4 fragments -> 8 v_wmma per 32-wide K step).
// amode/bmode/cmode: 0 = uniform batch stride, 1 = per-(b,h) head slice.
template <int BNN, int RELU>
__global__ __launch_bounds__(128) void gemm_wmma(
    const float* __restrict__ A, const float* __restrict__ B, float* __restrict__ C,
    const float* __restrict__ bias,
    int M, int N, int K, int lda, int ldb, int ldc,
    int64_t asb, int64_t bsb, int64_t csb,
    int amode, int bmode, int cmode,
    float alpha)
{
  int z = blockIdx.z;
  const float* Ab = A + (amode ? head_off(z, lda) : (int64_t)z * asb);
  const float* Bb = B + (bmode ? head_off(z, ldb) : (int64_t)z * bsb);
  float*       Cb = C + (cmode ? head_off(z, ldc) : (int64_t)z * csb);

  int wave  = threadIdx.x >> 5;
  int tileM = blockIdx.y * 128 + wave * 32;
  int tileN = blockIdx.x * 64;

  v8f acc[2][4] = {};

  for (int k0 = 0; k0 < K; k0 += 32) {
    // Prefetch next K tile of A rows (speculative; OOB silently dropped).
    {
      int lane = threadIdx.x & 31;
      const float* pf = Ab + (int64_t)(tileM + (lane & 15)) * lda + (k0 + 32);
      __builtin_prefetch(pf, 0, 1);
    }

    v16h a0 = load_frag_a(Ab, lda, tileM,      k0);
    v16h a1 = load_frag_a(Ab, lda, tileM + 16, k0);
    v16h b[4];
#pragma unroll
    for (int j = 0; j < 4; ++j) {
      b[j] = BNN ? load_frag_b_nn(Bb, ldb, tileN + 16 * j, k0)
                 : load_frag_b_nt(Bb, ldb, tileN + 16 * j, k0);
    }
#pragma unroll
    for (int j = 0; j < 4; ++j) {
      acc[0][j] = __builtin_amdgcn_wmma_f32_16x16x32_f16(false, a0, false, b[j], (short)0, acc[0][j], false, false);
      acc[1][j] = __builtin_amdgcn_wmma_f32_16x16x32_f16(false, a1, false, b[j], (short)0, acc[1][j], false, false);
    }
  }

  // C/D layout: VGPR r, lanes 0-15 -> M=r, N=lane; lanes 16-31 -> M=8+r.
  int lane  = threadIdx.x & 31;
  int cc    = lane & 15;
  int rbase = (lane >> 4) * 8;
#pragma unroll
  for (int i = 0; i < 2; ++i) {
    int rowT = tileM + i * 16 + rbase;
#pragma unroll
    for (int j = 0; j < 4; ++j) {
      int colT = tileN + j * 16 + cc;
      float bv = bias ? bias[colT] : 0.0f;
#pragma unroll
      for (int r = 0; r < 8; ++r) {
        float v = acc[i][j][r] * alpha + bv;
        if (RELU) v = v > 0.0f ? v : 0.0f;
        Cb[(int64_t)(rowT + r) * ldc + colT] = v;
      }
    }
  }
}

// Row softmax with key-padding (+ optional causal) mask, matching
// jnp.where(mask, s, -1e9) then softmax.  One block per (z, q) row.
__global__ __launch_bounds__(256) void masked_softmax(
    float* __restrict__ scores, const int* __restrict__ toks, int causal)
{
  __shared__ float red[256];
  int q = blockIdx.x;
  int z = blockIdx.y;
  int b = z / H_;
  float* row = scores + ((int64_t)z * S_ + q) * S_;
  int t = threadIdx.x;
  int c0 = t, c1 = t + 256;
  bool m0 = (toks[b * S_ + c0] != 0) && (!causal || c0 <= q);
  bool m1 = (toks[b * S_ + c1] != 0) && (!causal || c1 <= q);
  float v0 = m0 ? row[c0] : -1e9f;
  float v1 = m1 ? row[c1] : -1e9f;

  red[t] = fmaxf(v0, v1);
  __syncthreads();
  for (int s = 128; s > 0; s >>= 1) { if (t < s) red[t] = fmaxf(red[t], red[t + s]); __syncthreads(); }
  float mx = red[0];
  __syncthreads();

  v0 = __expf(v0 - mx);
  v1 = __expf(v1 - mx);
  red[t] = v0 + v1;
  __syncthreads();
  for (int s = 128; s > 0; s >>= 1) { if (t < s) red[t] += red[t + s]; __syncthreads(); }
  float inv = 1.0f / red[0];
  row[c0] = v0 * inv;
  row[c1] = v1 * inv;
}

// out = LayerNorm(x + r) * g + b   (one block per 512-wide row)
__global__ __launch_bounds__(256) void add_layernorm(
    const float* __restrict__ x, const float* __restrict__ r,
    const float* __restrict__ g, const float* __restrict__ bta,
    float* __restrict__ out)
{
  __shared__ float red[256];
  int row = blockIdx.x;
  int t = threadIdx.x;
  const float* xr = x + (int64_t)row * D_;
  const float* rr = r + (int64_t)row * D_;
  float v0 = xr[t] + rr[t];
  float v1 = xr[t + 256] + rr[t + 256];

  red[t] = v0 + v1;
  __syncthreads();
  for (int s = 128; s > 0; s >>= 1) { if (t < s) red[t] += red[t + s]; __syncthreads(); }
  float mean = red[0] * (1.0f / (float)D_);
  __syncthreads();

  float d0 = v0 - mean, d1 = v1 - mean;
  red[t] = d0 * d0 + d1 * d1;
  __syncthreads();
  for (int s = 128; s > 0; s >>= 1) { if (t < s) red[t] += red[t + s]; __syncthreads(); }
  float inv = rsqrtf(red[0] * (1.0f / (float)D_) + 1e-5f);

  float* o = out + (int64_t)row * D_;
  o[t]       = d0 * inv * g[t]       + bta[t];
  o[t + 256] = d1 * inv * g[t + 256] + bta[t + 256];
}

// x[b,s,:] = emb[tok] * sqrt(D) + pe[s,:]
__global__ __launch_bounds__(256) void embed_pe(
    const int* __restrict__ toks, const float* __restrict__ emb,
    const float* __restrict__ pe, float* __restrict__ out)
{
  int idx = blockIdx.x;           // b*S + s
  int s   = idx & (S_ - 1);
  int tok = toks[idx];
  int t   = threadIdx.x;
  const float* e = emb + (int64_t)tok * D_;
  const float* p = pe  + (int64_t)s * D_;
  float* o = out + (int64_t)idx * D_;
  o[t]       = e[t]       * EMB_SCALE_ + p[t];
  o[t + 256] = e[t + 256] * EMB_SCALE_ + p[t + 256];
}

// ---------------------------------------------------------------------------
// Host orchestration
// ---------------------------------------------------------------------------

static inline void gemm_nt(hipStream_t st, const float* A, const float* W,
                           const float* bias, float* C, int M, int N, int K,
                           int relu = 0)
{
  dim3 grid(N / 64, M / 128, 1);
  if (relu)
    gemm_wmma<0, 1><<<grid, 128, 0, st>>>(A, W, C, bias, M, N, K, K, K, N,
                                          0, 0, 0, 0, 0, 0, 1.0f);
  else
    gemm_wmma<0, 0><<<grid, 128, 0, st>>>(A, W, C, bias, M, N, K, K, K, N,
                                          0, 0, 0, 0, 0, 0, 1.0f);
}

static inline void attn_scores(hipStream_t st, const float* Q, const float* Kb, float* sc)
{
  dim3 grid(S_ / 64, S_ / 128, BH_);
  gemm_wmma<0, 0><<<grid, 128, 0, st>>>(Q, Kb, sc, nullptr, S_, S_, DK_,
                                        D_, D_, S_, 0, 0, (int64_t)S_ * S_,
                                        1, 1, 0, INV_SCALE_);
}

static inline void attn_pv(hipStream_t st, const float* P, const float* V, float* O)
{
  dim3 grid(DK_ / 64, S_ / 128, BH_);
  gemm_wmma<1, 0><<<grid, 128, 0, st>>>(P, V, O, nullptr, S_, DK_, S_,
                                        S_, D_, D_, (int64_t)S_ * S_, 0, 0,
                                        0, 1, 1, 1.0f);
}

extern "C" void kernel_launch(void* const* d_in, const int* in_sizes, int n_in,
                              void* d_out, int out_size, void* d_ws, size_t ws_size,
                              hipStream_t stream) {
  (void)in_sizes; (void)n_in; (void)out_size; (void)ws_size;

  // --- inputs (depth-first dict order of setup_inputs) ---
  const int*   src      = (const int*)d_in[0];
  const int*   tgt      = (const int*)d_in[1];
  const float* src_emb  = (const float*)d_in[2];
  const float* tgt_emb  = (const float*)d_in[3];
  const float* pe       = (const float*)d_in[4];
  // encoder
  const float* e_wq = (const float*)d_in[5];   const float* e_bq = (const float*)d_in[6];
  const float* e_wk = (const float*)d_in[7];   const float* e_bk = (const float*)d_in[8];
  const float* e_wv = (const float*)d_in[9];   const float* e_bv = (const float*)d_in[10];
  const float* e_wo = (const float*)d_in[11];  const float* e_bo = (const float*)d_in[12];
  const float* e_w1 = (const float*)d_in[13];  const float* e_b1 = (const float*)d_in[14];
  const float* e_w2 = (const float*)d_in[15];  const float* e_b2 = (const float*)d_in[16];
  const float* e_l1g = (const float*)d_in[17]; const float* e_l1b = (const float*)d_in[18];
  const float* e_l2g = (const float*)d_in[19]; const float* e_l2b = (const float*)d_in[20];
  // decoder self-attn
  const float* sa_wq = (const float*)d_in[21]; const float* sa_bq = (const float*)d_in[22];
  const float* sa_wk = (const float*)d_in[23]; const float* sa_bk = (const float*)d_in[24];
  const float* sa_wv = (const float*)d_in[25]; const float* sa_bv = (const float*)d_in[26];
  const float* sa_wo = (const float*)d_in[27]; const float* sa_bo = (const float*)d_in[28];
  // decoder cross-attn
  const float* ca_wq = (const float*)d_in[29]; const float* ca_bq = (const float*)d_in[30];
  const float* ca_wk = (const float*)d_in[31]; const float* ca_bk = (const float*)d_in[32];
  const float* ca_wv = (const float*)d_in[33]; const float* ca_bv = (const float*)d_in[34];
  const float* ca_wo = (const float*)d_in[35]; const float* ca_bo = (const float*)d_in[36];
  // decoder ff + norms
  const float* d_w1 = (const float*)d_in[37];  const float* d_b1 = (const float*)d_in[38];
  const float* d_w2 = (const float*)d_in[39];  const float* d_b2 = (const float*)d_in[40];
  const float* d_l1g = (const float*)d_in[41]; const float* d_l1b = (const float*)d_in[42];
  const float* d_l2g = (const float*)d_in[43]; const float* d_l2b = (const float*)d_in[44];
  const float* d_l3g = (const float*)d_in[45]; const float* d_l3b = (const float*)d_in[46];
  const float* out_w = (const float*)d_in[47]; const float* out_b = (const float*)d_in[48];

  float* logits = (float*)d_out;

  // --- workspace carve (floats) ---
  const int64_t ACT = (int64_t)NTOK_ * D_;        // 1,048,576
  float* ws   = (float*)d_ws;
  float* x    = ws;                 // encoder activation (persistent enc_out)
  float* xd   = x    + ACT;         // decoder activation
  float* y    = xd   + ACT;         // ln1 output
  float* t2   = y    + ACT;         // residual temp
  float* t3   = t2   + ACT;         // dec ln2 output
  float* qb   = t3   + ACT;
  float* kb   = qb   + ACT;
  float* vb   = kb   + ACT;
  float* attn = vb   + ACT;
  float* ffb  = attn + ACT;         // [2048, 2048]
  float* sc   = ffb  + (int64_t)NTOK_ * FF_;  // [32, 512, 512]

  const int64_t WDD = (int64_t)D_ * D_;
  const int64_t WFD = (int64_t)FF_ * D_;

  // --- embeddings ---
  embed_pe<<<NTOK_, 256, 0, stream>>>(src, src_emb, pe, x);
  embed_pe<<<NTOK_, 256, 0, stream>>>(tgt, tgt_emb, pe, xd);

  // --- encoder ---
  for (int l = 0; l < 6; ++l) {
    gemm_nt(stream, x, e_wq + l * WDD, e_bq + l * D_, qb, NTOK_, D_, D_);
    gemm_nt(stream, x, e_wk + l * WDD, e_bk + l * D_, kb, NTOK_, D_, D_);
    gemm_nt(stream, x, e_wv + l * WDD, e_bv + l * D_, vb, NTOK_, D_, D_);
    attn_scores(stream, qb, kb, sc);
    masked_softmax<<<dim3(S_, BH_), 256, 0, stream>>>(sc, src, 0);
    attn_pv(stream, sc, vb, attn);
    gemm_nt(stream, attn, e_wo + l * WDD, e_bo + l * D_, t2, NTOK_, D_, D_);
    add_layernorm<<<NTOK_, 256, 0, stream>>>(x, t2, e_l1g + l * D_, e_l1b + l * D_, y);
    gemm_nt(stream, y, e_w1 + l * WFD, e_b1 + l * FF_, ffb, NTOK_, FF_, D_, 1);
    gemm_nt(stream, ffb, e_w2 + l * WFD, e_b2 + l * D_, t2, NTOK_, D_, FF_);
    add_layernorm<<<NTOK_, 256, 0, stream>>>(y, t2, e_l2g + l * D_, e_l2b + l * D_, x);
  }

  // --- decoder ---
  for (int l = 0; l < 6; ++l) {
    // self-attention (causal + tgt pad mask)
    gemm_nt(stream, xd, sa_wq + l * WDD, sa_bq + l * D_, qb, NTOK_, D_, D_);
    gemm_nt(stream, xd, sa_wk + l * WDD, sa_bk + l * D_, kb, NTOK_, D_, D_);
    gemm_nt(stream, xd, sa_wv + l * WDD, sa_bv + l * D_, vb, NTOK_, D_, D_);
    attn_scores(stream, qb, kb, sc);
    masked_softmax<<<dim3(S_, BH_), 256, 0, stream>>>(sc, tgt, 1);
    attn_pv(stream, sc, vb, attn);
    gemm_nt(stream, attn, sa_wo + l * WDD, sa_bo + l * D_, t2, NTOK_, D_, D_);
    add_layernorm<<<NTOK_, 256, 0, stream>>>(xd, t2, d_l1g + l * D_, d_l1b + l * D_, y);

    // cross-attention (K/V from encoder output x, src pad mask)
    gemm_nt(stream, y, ca_wq + l * WDD, ca_bq + l * D_, qb, NTOK_, D_, D_);
    gemm_nt(stream, x, ca_wk + l * WDD, ca_bk + l * D_, kb, NTOK_, D_, D_);
    gemm_nt(stream, x, ca_wv + l * WDD, ca_bv + l * D_, vb, NTOK_, D_, D_);
    attn_scores(stream, qb, kb, sc);
    masked_softmax<<<dim3(S_, BH_), 256, 0, stream>>>(sc, src, 0);
    attn_pv(stream, sc, vb, attn);
    gemm_nt(stream, attn, ca_wo + l * WDD, ca_bo + l * D_, t2, NTOK_, D_, D_);
    add_layernorm<<<NTOK_, 256, 0, stream>>>(y, t2, d_l2g + l * D_, d_l2b + l * D_, t3);

    // feed-forward
    gemm_nt(stream, t3, d_w1 + l * WFD, d_b1 + l * FF_, ffb, NTOK_, FF_, D_, 1);
    gemm_nt(stream, ffb, d_w2 + l * WFD, d_b2 + l * D_, t2, NTOK_, D_, FF_);
    add_layernorm<<<NTOK_, 256, 0, stream>>>(t3, t2, d_l3g + l * D_, d_l3b + l * D_, xd);
  }

  // --- output projection: [2048,512] @ [32000,512]^T + out_b -> d_out ---
  gemm_nt(stream, xd, out_w, out_b, logits, NTOK_, TGTV_, D_);
}
